// SlotAttention_15874199126074
// MI455X (gfx1250) — matrix operand order repaired and checked by reading, not compiled, not run on hardware
//
#include <hip/hip_runtime.h>
#include <hip/hip_bf16.h>

// ---------------- CDNA5 WMMA types / helpers ----------------
typedef __attribute__((ext_vector_type(16))) _Float16 v16h;
typedef __attribute__((ext_vector_type(8)))  float    v8f;

union AF { v16h v; unsigned u[8]; };

__device__ __forceinline__ unsigned pack2h(float a, float b) {
  union { _Float16 h[2]; unsigned u; } t;
  t.h[0] = (_Float16)a; t.h[1] = (_Float16)b;
  return t.u;
}

// 16-bit A-matrix 16x32 layout: VGPR j, lane-half h -> base K
#define KA(j, h) ((((j) & 3) << 1) + ((h) << 3) + (((j) >> 2) << 4))
// contiguous K order (not used here): VGPR j, lane-half h -> base K
#define KB(j, h) (((j) << 1) + ((h) << 4))

#define BATCH 32
#define NTOK  4096
#define DIN   128
#define HD    64
#define NSLOT 8

// ---------------- slots = mu + (softplus(logsigma)+1e-5)*noise ----------------
__global__ void k_init_slots(const float* __restrict__ mu, const float* __restrict__ ls,
                             const float* __restrict__ noise, float* __restrict__ slots) {
  int idx = blockIdx.x * 256 + threadIdx.x;        // 32*8*64 = 16384
  int r = idx & 511;                               // broadcast over batch
  float l = ls[r];
  float sp = fmaxf(l, 0.f) + log1pf(expf(-fabsf(l))) + 1e-5f;
  slots[idx] = mu[r] + sp * noise[idx];
}

// ---------------- fused LayerNorm(x) + k/v projection (WMMA f16->f32) ----------------
// 128 threads = 4 waves; each wave owns a 16-row token tile; block does 64 rows.
__global__ __launch_bounds__(128) void k_ln_proj_kv(
    const float* __restrict__ x, const float* __restrict__ Wk, const float* __restrict__ Wv,
    const float* __restrict__ g, const float* __restrict__ bb,
    _Float16* __restrict__ kout, _Float16* __restrict__ vout) {
  __shared__ unsigned Wlds[2][64][64];   // [mat][out-ch][d-pair]  f16x2
  __shared__ unsigned XT[4][16][64];     // per-wave LN'd x tile   f16x2

  const int tid = threadIdx.x, wave = tid >> 5, lane = tid & 31;
  const int m = lane & 15, hh = lane >> 4;

  for (int i = tid; i < 4096; i += 128) {     // 64x128 per matrix, packed as pairs
    int ch = i >> 6, dp = i & 63;
    const float* pk = Wk + ch * 128 + dp * 2;
    const float* pv = Wv + ch * 128 + dp * 2;
    Wlds[0][ch][dp] = pack2h(pk[0], pk[1]);
    Wlds[1][ch][dp] = pack2h(pv[0], pv[1]);
  }

  long rowbase = (long)blockIdx.x * 64 + wave * 16;
  const float* xr = x + (rowbase + m) * (long)DIN + hh * 64;  // lane: row m, half hh
  float vals[64];
  float s = 0.f, s2 = 0.f;
#pragma unroll
  for (int i = 0; i < 64; ++i) { float t = xr[i]; vals[i] = t; s += t; s2 += t * t; }
  s  += __shfl_xor(s, 16, 32);
  s2 += __shfl_xor(s2, 16, 32);
  float mean = s * (1.f / 128.f);
  float var  = s2 * (1.f / 128.f) - mean * mean;
  float inv  = rsqrtf(var + 1e-5f);
#pragma unroll
  for (int i = 0; i < 64; i += 2) {
    int c0 = hh * 64 + i;
    float a = (vals[i]     - mean) * inv * g[c0]     + bb[c0];
    float c = (vals[i + 1] - mean) * inv * g[c0 + 1] + bb[c0 + 1];
    XT[wave][m][c0 >> 1] = pack2h(a, c);
  }
  __syncthreads();

  // out tile per wave: 16 tokens x (64 k-cols + 64 v-cols), K=128 contraction
#pragma unroll
  for (int c = 0; c < 8; ++c) {
    const int mat = c >> 2;
    const int cb  = (c & 3) * 16;
    v8f C = {};
#pragma unroll
    for (int s4 = 0; s4 < 4; ++s4) {   // 4 x K=32
      AF a, bf;
#pragma unroll
      for (int j = 0; j < 8; ++j) {
        a.u[j]  = XT[wave][m][16 * s4 + (KA(j, hh) >> 1)];
        bf.u[j] = Wlds[mat][cb + m][16 * s4 + (KA(j, hh) >> 1)];
      }
      C = __builtin_amdgcn_wmma_f32_16x16x32_f16(false, a.v, false, bf.v, (short)0, C, false, false);
    }
    _Float16* outp = mat ? vout : kout;
#pragma unroll
    for (int r = 0; r < 8; ++r) {
      long tok = rowbase + r + 8 * hh;           // D layout: M = r + 8*half
      outp[tok * HD + cb + m] = (_Float16)C[r];  // N = lane%16
    }
  }
}

// ---------------- per-iter: LN(slots), q = s@Wq^T (padded to 16 slots), zero accum ----------------
__global__ __launch_bounds__(64) void k_qprep(
    const float* __restrict__ slots, const float* __restrict__ g, const float* __restrict__ bb,
    const float* __restrict__ Wq, _Float16* __restrict__ qpad,
    float* __restrict__ colsum, float* __restrict__ updates) {
  int b = blockIdx.x >> 4, k = blockIdx.x & 15, h = threadIdx.x;
  if (k >= NSLOT) { qpad[(b * 16 + k) * HD + h] = (_Float16)0.f; return; }
  float sv = slots[(b * NSLOT + k) * HD + h];
  __shared__ float sb[64];
  sb[h] = sv; __syncthreads();
  float mean = 0.f;
  for (int i = 0; i < 64; ++i) mean += sb[i];
  mean *= (1.f / 64.f);
  float var = 0.f;
  for (int i = 0; i < 64; ++i) { float d = sb[i] - mean; var += d * d; }
  var *= (1.f / 64.f);
  float sl = (sv - mean) * rsqrtf(var + 1e-5f) * g[h] + bb[h];
  __shared__ float ss[64];
  ss[h] = sl; __syncthreads();
  float acc = 0.f;
  for (int i = 0; i < 64; ++i) acc += ss[i] * Wq[h * 64 + i];   // q[b,k,g=h]
  qpad[(b * 16 + k) * HD + h] = (_Float16)acc;
  updates[(b * NSLOT + k) * HD + h] = 0.f;
  if (h == 0) colsum[b * NSLOT + k] = 0.f;
}

// ---------------- scores = H^-0.5 * q@k^T tile, in-lane softmax over 8 slots ----------------
// grid: B * (N/128); 8 waves x 16 tokens. WMMA: M=16 slots(pad), N=16 tokens, K=64.
__global__ __launch_bounds__(256) void k_scores(
    const _Float16* __restrict__ kbuf, const _Float16* __restrict__ qpad,
    float* __restrict__ attn, float* __restrict__ colsum) {
  __shared__ unsigned qlds[16][32];
  int b = blockIdx.x >> 5, chunk = blockIdx.x & 31;
  int tid = threadIdx.x, wave = tid >> 5, lane = tid & 31;
  int m = lane & 15, hh = lane >> 4;
  const unsigned* qsrc = (const unsigned*)(qpad + (long)b * 16 * HD);
  for (int i = tid; i < 512; i += 256) qlds[i >> 5][i & 31] = qsrc[i];
  __syncthreads();

  long tokbase = (long)b * NTOK + chunk * 128 + wave * 16;
  const _Float16* kb = kbuf + tokbase * HD;
  v8f C = {};
#pragma unroll
  for (int s4 = 0; s4 < 2; ++s4) {
    AF a, bf;
#pragma unroll
    for (int j = 0; j < 8; ++j) {
      a.u[j]  = qlds[m][16 * s4 + (KA(j, hh) >> 1)];                         // A: q (slot m)
      bf.u[j] = *(const unsigned*)(kb + (long)m * HD + 32 * s4 + KA(j, hh)); // B: k^T (token m)
    }
    C = __builtin_amdgcn_wmma_f32_16x16x32_f16(false, a.v, false, bf.v, (short)0, C, false, false);
  }
  // lanes 0-15 own all 8 real slot scores of one token
  float sc[8];
  float mx = -1e30f;
#pragma unroll
  for (int r = 0; r < 8; ++r) { sc[r] = 0.125f * C[r]; mx = fmaxf(mx, sc[r]); }
  float sum = 0.f;
#pragma unroll
  for (int r = 0; r < 8; ++r) { sc[r] = expf(sc[r] - mx); sum += sc[r]; }
  float rs = 1.f / sum;
#pragma unroll
  for (int r = 0; r < 8; ++r) sc[r] = sc[r] * rs + 1e-8f;
  if (hh == 0) {
    float4* ap = (float4*)(attn + (tokbase + m) * NSLOT);
    ap[0] = make_float4(sc[0], sc[1], sc[2], sc[3]);
    ap[1] = make_float4(sc[4], sc[5], sc[6], sc[7]);
  }
#pragma unroll
  for (int r = 0; r < 8; ++r) {           // column sums over tokens (lanes 0-15)
    float t = sc[r];
    t += __shfl_xor(t, 8, 16);
    t += __shfl_xor(t, 4, 16);
    t += __shfl_xor(t, 2, 16);
    t += __shfl_xor(t, 1, 16);
    if (lane == 0) atomicAdd(&colsum[b * NSLOT + r], t);
  }
}

// ---------------- updates = (attn/colsum)^T @ v : M=16 slots(pad), N=16 h, K=tokens ----------------
// grid: B*4; 8 waves x 128 tokens each; accumulate to global with f32 atomics.
__global__ __launch_bounds__(256) void k_updates(
    const float* __restrict__ attn, const float* __restrict__ colsum,
    const _Float16* __restrict__ vbuf, float* __restrict__ updates) {
  int b = blockIdx.x >> 2, chunk = blockIdx.x & 3;
  int tid = threadIdx.x, wave = tid >> 5, lane = tid & 31;
  int m = lane & 15, hh = lane >> 4;
  int tokbase = chunk * 1024 + wave * 128;
  float invc = 0.f;
  if (m < NSLOT) invc = 1.f / colsum[b * NSLOT + m];
  const float*    ab = attn + (long)b * NTOK * NSLOT;
  const _Float16* vb = vbuf + (long)b * NTOK * HD;

  AF afr[4];
#pragma unroll
  for (int s4 = 0; s4 < 4; ++s4) {
#pragma unroll
    for (int j = 0; j < 8; ++j) {
      int t0 = tokbase + 32 * s4 + KA(j, hh);
      float a0 = 0.f, a1 = 0.f;
      if (m < NSLOT) {
        a0 = ab[(long)t0 * NSLOT + m] * invc;
        a1 = ab[(long)(t0 + 1) * NSLOT + m] * invc;
      }
      afr[s4].u[j] = pack2h(a0, a1);
    }
  }
#pragma unroll
  for (int ht = 0; ht < 4; ++ht) {
    int hb = ht * 16;
    v8f C = {};
#pragma unroll
    for (int s4 = 0; s4 < 4; ++s4) {
      AF bf;
#pragma unroll
      for (int j = 0; j < 8; ++j) {
        int t0 = tokbase + 32 * s4 + KA(j, hh);
        union { _Float16 h[2]; unsigned u; } t;
        t.h[0] = vb[(long)t0 * HD + hb + m];
        t.h[1] = vb[(long)(t0 + 1) * HD + hb + m];
        bf.u[j] = t.u;
      }
      C = __builtin_amdgcn_wmma_f32_16x16x32_f16(false, afr[s4].v, false, bf.v, (short)0, C, false, false);
    }
    if (hh == 0) {   // valid slot rows 0..7
#pragma unroll
      for (int r = 0; r < 8; ++r)
        atomicAdd(&updates[(b * NSLOT + r) * HD + hb + m], C[r]);
    }
  }
}

// ---------------- GRU cell + LN + MLP residual (tiny, f32 VALU) ----------------
__global__ __launch_bounds__(64) void k_slotupd(
    const float* __restrict__ updates, const float* __restrict__ slots_in,
    const float* __restrict__ wih, const float* __restrict__ whh,
    const float* __restrict__ bih, const float* __restrict__ bhh,
    const float* __restrict__ lng, const float* __restrict__ lnb,
    const float* __restrict__ w1, const float* __restrict__ b1,
    const float* __restrict__ w2, const float* __restrict__ b2,
    float* __restrict__ out) {
  int bk = blockIdx.x, h = threadIdx.x;
  __shared__ float su[64], shp[64], sh1[64], ssl[64], shid[128];
  float u  = updates[bk * 64 + h];
  float hp = slots_in[bk * 64 + h];
  su[h] = u; shp[h] = hp; __syncthreads();
  float gr = bih[h], gz = bih[64 + h], gn = bih[128 + h];
  float hr = bhh[h], hz = bhh[64 + h], hn = bhh[128 + h];
  for (int i = 0; i < 64; ++i) {
    float ui = su[i], hi = shp[i];
    gr += ui * wih[h * 64 + i];         gz += ui * wih[(64 + h) * 64 + i];
    gn += ui * wih[(128 + h) * 64 + i];
    hr += hi * whh[h * 64 + i];         hz += hi * whh[(64 + h) * 64 + i];
    hn += hi * whh[(128 + h) * 64 + i];
  }
  float r = 1.f / (1.f + expf(-(gr + hr)));
  float z = 1.f / (1.f + expf(-(gz + hz)));
  float n = tanhf(gn + r * hn);
  float hnew = (1.f - z) * n + z * hp;
  sh1[h] = hnew; __syncthreads();
  float mean = 0.f; for (int i = 0; i < 64; ++i) mean += sh1[i]; mean *= (1.f / 64.f);
  float var = 0.f;  for (int i = 0; i < 64; ++i) { float d = sh1[i] - mean; var += d * d; }
  var *= (1.f / 64.f);
  float sl = (hnew - mean) * rsqrtf(var + 1e-5f) * lng[h] + lnb[h];
  ssl[h] = sl; __syncthreads();
  float h0 = b1[h], h1v = b1[64 + h];
  for (int i = 0; i < 64; ++i) { float s = ssl[i]; h0 += s * w1[h * 64 + i]; h1v += s * w1[(64 + h) * 64 + i]; }
  shid[h] = fmaxf(h0, 0.f); shid[64 + h] = fmaxf(h1v, 0.f); __syncthreads();
  float o = b2[h];
  for (int i = 0; i < 128; ++i) o += shid[i] * w2[h * 128 + i];
  out[bk * 64 + h] = sl + o;
}

// ---------------- host side ----------------
extern "C" void kernel_launch(void* const* d_in, const int* in_sizes, int n_in,
                              void* d_out, int out_size, void* d_ws, size_t ws_size,
                              hipStream_t stream) {
  (void)in_sizes; (void)n_in; (void)out_size; (void)ws_size;
  const float* x        = (const float*)d_in[0];
  const float* noise    = (const float*)d_in[1];
  const float* Wq       = (const float*)d_in[2];
  const float* Wk       = (const float*)d_in[3];
  const float* Wv       = (const float*)d_in[4];
  const float* ln_in_g  = (const float*)d_in[5];
  const float* ln_in_b  = (const float*)d_in[6];
  const float* ln_sl_g  = (const float*)d_in[7];
  const float* ln_sl_b  = (const float*)d_in[8];
  const float* ln_mlp_g = (const float*)d_in[9];
  const float* ln_mlp_b = (const float*)d_in[10];
  const float* gru_w_ih = (const float*)d_in[11];
  const float* gru_w_hh = (const float*)d_in[12];
  const float* gru_b_ih = (const float*)d_in[13];
  const float* gru_b_hh = (const float*)d_in[14];
  const float* mlp_w1   = (const float*)d_in[15];
  const float* mlp_b1   = (const float*)d_in[16];
  const float* mlp_w2   = (const float*)d_in[17];
  const float* mlp_b2   = (const float*)d_in[18];
  const float* mu       = (const float*)d_in[19];
  const float* lsg      = (const float*)d_in[20];

  char* ws = (char*)d_ws;
  const size_t KV = (size_t)BATCH * NTOK * HD * sizeof(_Float16);  // 16 MB each
  _Float16* kbuf  = (_Float16*)(ws);
  _Float16* vbuf  = (_Float16*)(ws + KV);
  float*    attn  = (float*)(ws + 2 * KV);                          // 4 MB
  size_t off = 2 * KV + (size_t)BATCH * NTOK * NSLOT * sizeof(float);
  _Float16* qpad  = (_Float16*)(ws + off);  off += (size_t)BATCH * 16 * HD * sizeof(_Float16);
  float*    slots = (float*)(ws + off);     off += (size_t)BATCH * NSLOT * HD * sizeof(float);
  float*    colsum= (float*)(ws + off);     off += (size_t)BATCH * NSLOT * sizeof(float);
  float*    upd   = (float*)(ws + off);

  k_init_slots<<<64, 256, 0, stream>>>(mu, lsg, noise, slots);
  k_ln_proj_kv<<<(BATCH * NTOK) / 64, 128, 0, stream>>>(x, Wk, Wv, ln_in_g, ln_in_b, kbuf, vbuf);

  for (int it = 0; it < 3; ++it) {
    k_qprep<<<BATCH * 16, 64, 0, stream>>>(slots, ln_sl_g, ln_sl_b, Wq, qpad, colsum, upd);
    k_scores<<<BATCH * (NTOK / 128), 256, 0, stream>>>(kbuf, qpad, attn, colsum);
    k_updates<<<BATCH * 4, 256, 0, stream>>>(attn, colsum, vbuf, upd);
    float* outp = (it == 2) ? (float*)d_out : slots;
    k_slotupd<<<BATCH * NSLOT, 64, 0, stream>>>(upd, slots, gru_w_ih, gru_w_hh, gru_b_ih, gru_b_hh,
                                                ln_mlp_g, ln_mlp_b, mlp_w1, mlp_b1, mlp_w2, mlp_b2, outp);
  }
}
